// Attention_57346403336225
// MI455X (gfx1250) — matrix operand order repaired and checked by reading, not compiled, not run on hardware
//
#include <hip/hip_runtime.h>
#include <hip/hip_bf16.h>
#include <math.h>

// ---------------------------------------------------------------------------
// MI455X (gfx1250) fused LSA attention, bf16 WMMA (f32 accumulate).
// B=16, N=1024, E=128 (=head dim), H=8, INNER=1024.
// Pipeline: prep_weights -> ln_qkv -> attn (flash, online softmax) -> proj.
// Workspace layout (bytes), total ~129 MB:
//   [0)          WqkvT  bf16 [3072][128]   786432
//   [786432)     WpT    bf16 [128][1024]   262144
//   [1048576)    Q      bf16 [128bh][1024][128]  33554432
//   [34603008)   K      bf16 [128bh][1024][128]  33554432
//   [68157440)   Vt     bf16 [128bh][128][1024]  33554432
//   [101711872)  O      bf16 [16384][1024]       33554432
// ---------------------------------------------------------------------------

typedef __attribute__((ext_vector_type(16))) __bf16 v16bf;
typedef __attribute__((ext_vector_type(8)))  __bf16 v8bf;
typedef __attribute__((ext_vector_type(8)))  float  v8f;

#define WAVE_WAIT_DS() asm volatile("s_wait_dscnt 0" ::: "memory")

__device__ __forceinline__ v8f wmma_bf16(v16bf a, v16bf b, v8f c) {
  return __builtin_amdgcn_wmma_f32_16x16x32_bf16(
      /*neg_a=*/false, a, /*neg_b=*/false, b,
      /*c_mod=*/(short)0, c, /*reuse_a=*/false, /*reuse_b=*/false);
}

// A fragment (16x32, bf16, row-major source with leading dim ld elements).
// Lane l holds row m=l&15; VGPRs 0-3: K = koff+{0..7},
// VGPRs 4-7: K = koff+8+{0..7}, koff = (l>=16)?16:0. Two 16B loads/lane.
__device__ __forceinline__ v16bf load_a_frag(const __bf16* src, int ld, int k0) {
  int l = threadIdx.x & 31;
  int m = l & 15;
  int koff = (l >> 4) ? 16 : 0;
  const __bf16* p0 = src + (size_t)m * ld + k0 + koff;
  v8bf lo = *(const v8bf*)p0;
  v8bf hi = *(const v8bf*)(p0 + 8);
  v16bf r;
#pragma unroll
  for (int i = 0; i < 8; ++i) { r[i] = lo[i]; r[8 + i] = hi[i]; }
  return r;
}

// B fragment (32x16, bf16) loaded from B^T stored row-major (srcT[n][k]).
// Lane l holds col n=l&15; lanes 0-15: K=0..15, lanes 16-31: K=16..31,
// contiguous -> one 32B load per lane.
__device__ __forceinline__ v16bf load_b_frag_T(const __bf16* srcT, int ld, int k0) {
  int l = threadIdx.x & 31;
  int n = l & 15;
  int khalf = (l >> 4) ? 16 : 0;
  return *(const v16bf*)(srcT + (size_t)n * ld + k0 + khalf);
}

// Reduce across the 16 lanes of each wave32 half (rows of the C layout).
__device__ __forceinline__ float half_max(float v) {
  v = fmaxf(v, __shfl_xor(v, 1, 32));
  v = fmaxf(v, __shfl_xor(v, 2, 32));
  v = fmaxf(v, __shfl_xor(v, 4, 32));
  v = fmaxf(v, __shfl_xor(v, 8, 32));
  return v;
}
__device__ __forceinline__ float half_sum(float v) {
  v += __shfl_xor(v, 1, 32);
  v += __shfl_xor(v, 2, 32);
  v += __shfl_xor(v, 4, 32);
  v += __shfl_xor(v, 8, 32);
  return v;
}

// ---------------------------------------------------------------------------
// K0: transpose + bf16-convert both weight matrices.
__global__ void prep_weights(const float* __restrict__ wqkv,
                             const float* __restrict__ wproj,
                             __bf16* __restrict__ wqkvT,
                             __bf16* __restrict__ wpT) {
  int idx = blockIdx.x * 256 + threadIdx.x;
  if (idx < 3072 * 128) {                 // wqkvT[c][e], src wqkv[e][c]
    int c = idx >> 7, e = idx & 127;
    wqkvT[idx] = (__bf16)wqkv[(size_t)e * 3072 + c];
  }
  if (idx < 128 * 1024) {                 // wpT[c][i], src wproj[i][c]
    int c = idx >> 10, i = idx & 1023;
    wpT[idx] = (__bf16)wproj[(size_t)i * 128 + c];
  }
}

// ---------------------------------------------------------------------------
// K1: LayerNorm + QKV GEMM. Block = 128 thr (4 waves) = 16 patches x 64 cols.
// grid = (16384/16, 3072/64) = (1024, 48).
__global__ __launch_bounds__(128) void ln_qkv(const float* __restrict__ x,
                                              const float* __restrict__ lnw,
                                              const float* __restrict__ lnb,
                                              const __bf16* __restrict__ wqkvT,
                                              __bf16* __restrict__ Q,
                                              __bf16* __restrict__ K,
                                              __bf16* __restrict__ Vt) {
  __shared__ __align__(32) __bf16 xn[16 * 136];       // 16 rows, padded ld=136
  __shared__ __align__(32) float  stage[4][16 * 20];  // per-wave transpose tile

  int tid = threadIdx.x, w = tid >> 5, l = tid & 31;
  int rt = blockIdx.x;            // row tile (16 patches)
  int c0 = blockIdx.y * 64;       // first output column of the block

  // ---- LayerNorm: 8 threads per row, 16 elems per thread ----
  {
    int r = tid >> 3, g = tid & 7;
    const float* xr = x + ((size_t)rt * 16 + r) * 128 + g * 16;
    float v[16], s = 0.f, ss = 0.f;
#pragma unroll
    for (int i = 0; i < 16; ++i) { v[i] = xr[i]; s += v[i]; ss += v[i] * v[i]; }
    s  += __shfl_xor(s, 1, 32);  s  += __shfl_xor(s, 2, 32);  s  += __shfl_xor(s, 4, 32);
    ss += __shfl_xor(ss, 1, 32); ss += __shfl_xor(ss, 2, 32); ss += __shfl_xor(ss, 4, 32);
    float mean = s * (1.f / 128.f);
    float var  = ss * (1.f / 128.f) - mean * mean;
    float rstd = rsqrtf(var + 1e-5f);
#pragma unroll
    for (int i = 0; i < 16; ++i) {
      int e = g * 16 + i;
      xn[r * 136 + e] = (__bf16)((v[i] - mean) * rstd * lnw[e] + lnb[e]);
    }
  }
  __syncthreads();

  // ---- GEMM: each wave one 16x16 tile, K=128 in 4 WMMA steps ----
  int tc = c0 + w * 16;
  const __bf16* Bt = wqkvT + (size_t)tc * 128;
  v8f acc = {};
#pragma unroll
  for (int c = 0; c < 4; ++c)
    acc = wmma_bf16(load_a_frag(xn, 136, 32 * c), load_b_frag_T(Bt, 128, 32 * c), acc);

  // ---- epilogue: scatter to Q/K (row-major) or Vt (transposed) ----
  int cl = l & 15, half8 = (l >> 4) * 8;
  int region = tc >> 10;          // 0=Q 1=K 2=V (64-col blocks never straddle)
  int cr = tc & 1023, h = cr >> 7, d0 = cr & 127;
  int b = rt >> 6, nb = (rt & 63) * 16;

  if (region < 2) {
    float* st = stage[w];
#pragma unroll
    for (int j = 0; j < 8; ++j) st[(half8 + j) * 20 + cl] = acc[j];
    WAVE_WAIT_DS();
    if (l < 16) {
      __bf16* dst = (region == 0 ? Q : K) +
                    (((size_t)(b * 8 + h) * 1024 + nb + l)) * 128 + d0;
      v8bf p0, p1;
#pragma unroll
      for (int i = 0; i < 8; ++i) {
        p0[i] = (__bf16)st[l * 20 + i];
        p1[i] = (__bf16)st[l * 20 + 8 + i];
      }
      *(v8bf*)dst = p0;
      *(v8bf*)(dst + 8) = p1;
    }
  } else {
    // C-layout rows (patches) are contiguous along n in Vt -> 16B store/lane.
    __bf16* dst = Vt + ((size_t)(b * 8 + h) * 128 + d0 + cl) * 1024 + nb + half8;
    v8bf pk;
#pragma unroll
    for (int j = 0; j < 8; ++j) pk[j] = (__bf16)acc[j];
    *(v8bf*)dst = pk;
  }
}

// ---------------------------------------------------------------------------
// K2: flash attention. Block = 256 thr (8 waves), wave = 16 query rows.
// grid = (1024/128, B*H) = (8, 128).
__global__ __launch_bounds__(256) void attn(const __bf16* __restrict__ Q,
                                            const __bf16* __restrict__ K,
                                            const __bf16* __restrict__ Vt,
                                            const float* __restrict__ scale,
                                            __bf16* __restrict__ O) {
  __shared__ __align__(32) __bf16 pbuf[8][16 * 40];  // per-wave P (16x32) tile
  __shared__ __align__(32) float  tbuf[8][16 * 20];  // per-wave O transpose tile

  int tid = threadIdx.x, w = tid >> 5, l = tid & 31;
  int cl = l & 15, half8 = (l >> 4) * 8;
  int bh = blockIdx.y, b = bh >> 3, h = bh & 7;
  int m0 = blockIdx.x * 128 + w * 16;                // query-row base (0..1008)

  const __bf16* Qb = Q  + (size_t)bh * 1024 * 128;
  const __bf16* Kb = K  + (size_t)bh * 1024 * 128;
  const __bf16* Vb = Vt + (size_t)bh * 128 * 1024;
  float sc = scale[h];

  v16bf qa[4];
#pragma unroll
  for (int c = 0; c < 4; ++c)
    qa[c] = load_a_frag(Qb + (size_t)m0 * 128, 128, 32 * c);

  v8f o[8];
#pragma unroll
  for (int dt = 0; dt < 8; ++dt) o[dt] = (v8f){};
  float mi[8], li[8];
#pragma unroll
  for (int j = 0; j < 8; ++j) { mi[j] = -INFINITY; li[j] = 0.f; }

  for (int kb = 0; kb < 1024; kb += 32) {
    // S = Q K^T for 16 rows x 32 keys (two 16x16 C tiles).
    v8f s0 = {}, s1 = {};
#pragma unroll
    for (int c = 0; c < 4; ++c) {
      v16bf kb0 = load_b_frag_T(Kb + (size_t)kb * 128, 128, 32 * c);
      v16bf kb1 = load_b_frag_T(Kb + (size_t)(kb + 16) * 128, 128, 32 * c);
      s0 = wmma_bf16(qa[c], kb0, s0);
      s1 = wmma_bf16(qa[c], kb1, s1);
    }

    // Scale + diagonal mask + online softmax (rows live in 16-lane halves).
    float fac[8];
#pragma unroll
    for (int j = 0; j < 8; ++j) {
      int row = m0 + half8 + j;
      float a0 = (row == kb + cl)      ? -3.0e38f : s0[j] * sc;
      float a1 = (row == kb + 16 + cl) ? -3.0e38f : s1[j] * sc;
      float bm = half_max(fmaxf(a0, a1));
      float mn = fmaxf(mi[j], bm);
      float f  = __expf(mi[j] - mn);
      float p0 = __expf(a0 - mn);
      float p1 = __expf(a1 - mn);
      li[j] = li[j] * f + half_sum(p0 + p1);
      mi[j] = mn;
      fac[j] = f;
      pbuf[w][(half8 + j) * 40 + cl]      = (__bf16)p0;
      pbuf[w][(half8 + j) * 40 + 16 + cl] = (__bf16)p1;
    }
#pragma unroll
    for (int dt = 0; dt < 8; ++dt)
#pragma unroll
      for (int j = 0; j < 8; ++j) o[dt][j] *= fac[j];

    WAVE_WAIT_DS();  // wave-local P tile visible (per-wave region, no barrier)
    v16bf pa = load_a_frag(&pbuf[w][0], 40, 0);
#pragma unroll
    for (int dt = 0; dt < 8; ++dt) {
      v16bf vb = load_b_frag_T(Vb + (size_t)(dt * 16) * 1024, 1024, kb);
      o[dt] = wmma_bf16(pa, vb, o[dt]);
    }
  }

  // Normalize and store O row-major [b*1024+n][h*128+d] via LDS transpose.
  float inv[8];
#pragma unroll
  for (int j = 0; j < 8; ++j) inv[j] = 1.0f / li[j];
#pragma unroll
  for (int dt = 0; dt < 8; ++dt) {
#pragma unroll
    for (int j = 0; j < 8; ++j) tbuf[w][(half8 + j) * 20 + cl] = o[dt][j] * inv[j];
    WAVE_WAIT_DS();
    if (l < 16) {
      __bf16* dst = O + ((size_t)b * 1024 + m0 + l) * 1024 + h * 128 + dt * 16;
      v8bf q0, q1;
#pragma unroll
      for (int i = 0; i < 8; ++i) {
        q0[i] = (__bf16)tbuf[w][l * 20 + i];
        q1[i] = (__bf16)tbuf[w][l * 20 + 8 + i];
      }
      *(v8bf*)dst = q0;
      *(v8bf*)(dst + 8) = q1;
    }
  }
}

// ---------------------------------------------------------------------------
// K3: output projection [16384,1024] x WpT + bias -> f32 out.
// Block = 128 thr (4 waves) = 16 patches x 64 cols; grid = (1024, 2).
__global__ __launch_bounds__(128) void proj(const __bf16* __restrict__ O,
                                            const __bf16* __restrict__ wpT,
                                            const float* __restrict__ bias,
                                            float* __restrict__ out) {
  int tid = threadIdx.x, w = tid >> 5, l = tid & 31;
  int rt = blockIdx.x;
  int c0 = blockIdx.y * 64 + w * 16;
  const __bf16* Arow = O + (size_t)rt * 16 * 1024;
  const __bf16* Bt   = wpT + (size_t)c0 * 1024;

  v8f acc = {};
#pragma unroll 4
  for (int k0 = 0; k0 < 1024; k0 += 32)
    acc = wmma_bf16(load_a_frag(Arow, 1024, k0), load_b_frag_T(Bt, 1024, k0), acc);

  int cl = l & 15, half8 = (l >> 4) * 8;
  float bv = bias[c0 + cl];
#pragma unroll
  for (int j = 0; j < 8; ++j)
    out[((size_t)rt * 16 + half8 + j) * 128 + c0 + cl] = acc[j] + bv;
}

// ---------------------------------------------------------------------------
extern "C" void kernel_launch(void* const* d_in, const int* in_sizes, int n_in,
                              void* d_out, int out_size, void* d_ws, size_t ws_size,
                              hipStream_t stream) {
  (void)in_sizes; (void)n_in; (void)out_size; (void)ws_size;
  const float* x     = (const float*)d_in[0];
  const float* lnw   = (const float*)d_in[1];
  const float* lnb   = (const float*)d_in[2];
  const float* wqkv  = (const float*)d_in[3];
  const float* scale = (const float*)d_in[4];
  const float* wproj = (const float*)d_in[5];
  const float* bproj = (const float*)d_in[6];
  float* out = (float*)d_out;

  char* ws = (char*)d_ws;
  __bf16* WqkvT = (__bf16*)(ws);
  __bf16* WpT   = (__bf16*)(ws + 786432);
  __bf16* Qw    = (__bf16*)(ws + 1048576);
  __bf16* Kw    = (__bf16*)(ws + 34603008);
  __bf16* Vtw   = (__bf16*)(ws + 68157440);
  __bf16* Ow    = (__bf16*)(ws + 101711872);

  prep_weights<<<1536, 256, 0, stream>>>(wqkv, wproj, WqkvT, WpT);
  ln_qkv<<<dim3(1024, 48), 128, 0, stream>>>(x, lnw, lnb, WqkvT, Qw, Kw, Vtw);
  attn<<<dim3(8, 128), 256, 0, stream>>>(Qw, Kw, Vtw, scale, Ow);
  proj<<<dim3(1024, 2), 128, 0, stream>>>(Ow, WpT, bproj, out);
}
